// HardMemory_39204461478015
// MI455X (gfx1250) — matrix operand order repaired and checked by reading, not compiled, not run on hardware
//
#include <hip/hip_runtime.h>
#include <hip/hip_bf16.h>
#include <math.h>

typedef __attribute__((ext_vector_type(16))) _Float16 v16h;
typedef __attribute__((ext_vector_type(8)))  _Float16 v8h;
typedef __attribute__((ext_vector_type(8)))  float    v8f;
typedef __attribute__((ext_vector_type(4)))  unsigned int u32x4;
typedef __attribute__((ext_vector_type(4)))  int      i32x4;
typedef __attribute__((ext_vector_type(8)))  int      i32x8;

#define C_DIM        512
#define HW           4096               // 64*64
#define ROWS_PER_WG  128
#define THREADS      256
#define KT_N         16                 // 512 / 32 k-tiles
#define TILE_HALVES  (KT_N * 32 * 16)   // 8192 f16 = 16 KB per B m-tile
#define PAIR_HALVES  (2 * TILE_HALVES)  // 16384 f16 = 32 KB per TDM transfer
#define THRESH       0.8f
#define EPSN         1e-12f

#if defined(__has_builtin)
#if __has_builtin(__builtin_amdgcn_tensor_load_to_lds) && \
    __has_builtin(__builtin_amdgcn_s_wait_tensorcnt)
#define HAVE_TDM 1
#endif
#endif

// ---------------------------------------------------------------------------
// Pre-pass: L2-normalize memory rows (fp32) and scatter into WMMA B-fragment
// layout as f16:  bws[ ((mt*16 + kt)*32 + lane)*16 + h ]
// B 32x16 k order used here (contiguous k, not AMD's table order): n = lane & 15 ; k = kt*32 + (lane<16 ? h : 16+h)
// ---------------------------------------------------------------------------
__global__ void hm_norm_swizzle_mem(const float* __restrict__ mem,
                                    _Float16* __restrict__ bws) {
    const int m = blockIdx.x;          // memory row, 0..M-1
    const int t = threadIdx.x;         // 64 threads
    const float* row = mem + (size_t)m * C_DIM;

    float v[8];
    float ss = 0.f;
#pragma unroll
    for (int j = 0; j < 8; ++j) {
        v[j] = row[t * 8 + j];
        ss += v[j] * v[j];
    }
#pragma unroll
    for (int off = 16; off >= 1; off >>= 1) ss += __shfl_xor(ss, off, 32);
    __shared__ float sp[2];
    if ((t & 31) == 0) sp[t >> 5] = ss;
    __syncthreads();
    const float rinv = 1.f / fmaxf(sqrtf(sp[0] + sp[1]), EPSN);

    const int mt = m >> 4;
    const int lp = m & 15;
#pragma unroll
    for (int j = 0; j < 8; ++j) {
        const int k    = t * 8 + j;
        const int kt   = k >> 5;
        const int r    = k & 31;
        const int lane = lp + ((r & 16) ? 16 : 0);
        const int h    = r & 15;
        bws[(((size_t)mt * KT_N + kt) * 32 + lane) * 16 + h] =
            (_Float16)(v[j] * rinv);
    }
}

#ifdef HAVE_TDM
// ---------------------------------------------------------------------------
// TDM: one tensor_load_to_lds moving a contiguous 32 KB B tile-pair into LDS.
// D# packing per CDNA5 ISA 8.3/8.4: group0 {count=1, lds_addr, global_addr,
// type=2}; group1 {data_size=2B, tensor_dim0=tile_dim0=16384, 1 row}.
// ---------------------------------------------------------------------------
__device__ __forceinline__ void tdm_load_b_pair(const _Float16* gsrc,
                                                unsigned int lds_off) {
    const unsigned long long ga = (unsigned long long)(size_t)gsrc;
    u32x4 g0;
    g0[0] = 1u;                                   // count=1 (valid user desc)
    g0[1] = lds_off;                              // lds_addr (bytes)
    g0[2] = (unsigned int)ga;                     // global_addr[31:0]
    g0[3] = ((unsigned int)(ga >> 32) & 0x01FFFFFFu) | (2u << 30); // [56:32]|type=2
    i32x8 g1;
    g1[0] = (int)(1u << 16);                      // data_size=1 -> 2 bytes
    g1[1] = (int)((PAIR_HALVES & 0xFFFFu) << 16); // tensor_dim0 lo -> bits[63:48]
    g1[2] = (int)(1u << 16);                      // tensor_dim0 hi=0, tensor_dim1=1
    g1[3] = (int)((PAIR_HALVES & 0xFFFFu) << 16); // tile_dim0 -> bits[127:112]
    g1[4] = 1;                                    // tile_dim1=1, tile_dim2=0
    g1[5] = (int)PAIR_HALVES;                     // tensor_dim0_stride lo
    g1[6] = (int)((PAIR_HALVES & 0xFFFFu) << 16); // stride0 hi=0, stride1 lo
    g1[7] = 0;                                    // stride1 hi
    i32x4 z4 = (i32x4)0;
#if __clang_major__ >= 23
    i32x8 z8 = (i32x8)0;
    __builtin_amdgcn_tensor_load_to_lds(g0, g1, z4, z4, z8, 0);
#else
    __builtin_amdgcn_tensor_load_to_lds(g0, g1, z4, z4, 0);
#endif
}
#endif

// ---------------------------------------------------------------------------
// Main kernel: per-WG 128 rows x full M sweep, two m-tiles per iteration.
//   sim[n, m] = <x[n,:], mem_norm[m,:]>  (f16 inputs, f32 accum via WMMA)
//   out[n, c] = (max_m sim / ||x[n]|| > 0.8) ? memory[argmax][c] : 0
// B tile-pairs staged once per WG into double-buffered LDS via TDM async DMA;
// two independent WMMA accumulator chains give the scheduler load/WMMA overlap.
// ---------------------------------------------------------------------------
__global__ __launch_bounds__(THREADS)
void hm_gemm_argmax_gather(const float* __restrict__ x,
                           const float* __restrict__ mem,
                           const _Float16* __restrict__ bws,
                           float* __restrict__ out,
                           int mTiles) {
    __shared__ __align__(16) _Float16 sA[ROWS_PER_WG * 32];    // 8 KB staging
    __shared__ __align__(32) _Float16 sB[2][PAIR_HALVES];      // 2 x 32 KB
    __shared__ float sRed[THREADS];
    __shared__ float sNorm[ROWS_PER_WG];
    __shared__ int   sIdx[ROWS_PER_WG];

    const int t    = threadIdx.x;
    const int w    = t >> 5;                 // wave 0..7
    const int lane = t & 31;

    const int grow0 = blockIdx.x * ROWS_PER_WG;   // global row base
    const int b     = grow0 >> 12;                // / HW
    const int n0    = grow0 & (HW - 1);
    const float* xb = x + (size_t)b * C_DIM * HW + n0;   // + c*HW + rowLocal

    // ---- Phase 1: load A tile (fp32 -> f16) via LDS, build register frags,
    //      accumulate per-row sum of squares for ||x||.
    const int lr   = t & 127;       // staging row 0..127
    const int cpar = t >> 7;        // c parity 0/1
    const int rowA = w * 16 + (lane & 15);          // A frag row (this wave)
    const int base = (lane < 16) ? 0 : 16;          // A frag k base, contiguous k = 16h + s (this code's order)

    float ss = 0.f;
    v16h a_frag[KT_N];

#pragma unroll
    for (int kc = 0; kc < KT_N; ++kc) {
#pragma unroll
        for (int j = 0; j < 16; ++j) {
            const int cl = 2 * j + cpar;                       // 0..31
            const float v = xb[(size_t)(kc * 32 + cl) * HW + lr];
            ss += v * v;
            sA[lr * 32 + cl] = (_Float16)v;
        }
        __syncthreads();
        const v8h lo = *(const v8h*)&sA[rowA * 32 + base];
        const v8h hi = *(const v8h*)&sA[rowA * 32 + base + 8];
        v16h f;
#pragma unroll
        for (int i = 0; i < 8; ++i) { f[i] = lo[i]; f[i + 8] = hi[i]; }
        a_frag[kc] = f;
        __syncthreads();
    }

    sRed[t] = ss;
    __syncthreads();
    if (t < ROWS_PER_WG) sNorm[t] = sqrtf(sRed[t] + sRed[t + ROWS_PER_WG]);

    // ---- Phase 2: sweep memory tile-pairs; B double-buffered through LDS.
    float rmax[8];
    int   ridx[8];
#pragma unroll
    for (int j = 0; j < 8; ++j) { rmax[j] = -3.4e38f; ridx[j] = 0; }

    const int mPairs = mTiles >> 1;          // two 16-col tiles per iteration

#ifdef HAVE_TDM
    if (w == 0) {
        tdm_load_b_pair(bws, (unsigned int)(size_t)&sB[0][0]);
        __builtin_amdgcn_s_wait_tensorcnt(0);
    }
    __syncthreads();
#else
    {
        const u32x4* g = (const u32x4*)bws;
        u32x4*       s = (u32x4*)&sB[0][0];
#pragma unroll
        for (int i = 0; i < 8; ++i) s[t + i * THREADS] = g[t + i * THREADS];
    }
    __syncthreads();
#endif

    int cur = 0;
    for (int mp = 0; mp < mPairs; ++mp) {
#ifdef HAVE_TDM
        if (w == 0 && mp + 1 < mPairs)
            tdm_load_b_pair(bws + (size_t)(mp + 1) * PAIR_HALVES,
                            (unsigned int)(size_t)&sB[cur ^ 1][0]);
#else
        u32x4 nb[8];
        if (mp + 1 < mPairs) {
            const u32x4* g = (const u32x4*)(bws + (size_t)(mp + 1) * PAIR_HALVES);
#pragma unroll
            for (int i = 0; i < 8; ++i) nb[i] = g[t + i * THREADS];
        }
#endif
        // Two independent WMMA chains over the same A fragments; their LDS
        // fragment loads interleave so neither chain waits for dscnt==0.
        v8f acc0 = {};
        v8f acc1 = {};
        const _Float16* bL = &sB[cur][lane * 16];
#pragma unroll
        for (int kt = 0; kt < KT_N; ++kt) {
            const v16h bf0 = *(const v16h*)(bL + kt * 512);
            const v16h bf1 = *(const v16h*)(bL + kt * 512 + TILE_HALVES);
            acc0 = __builtin_amdgcn_wmma_f32_16x16x32_f16(
                false, a_frag[kt], false, bf0, (short)0, acc0, false, false);
            acc1 = __builtin_amdgcn_wmma_f32_16x16x32_f16(
                false, a_frag[kt], false, bf1, (short)0, acc1, false, false);
        }

        const int col0 = (mp * 2) * 16 + (lane & 15);
#pragma unroll
        for (int j = 0; j < 8; ++j) {
            if (acc0[j] > rmax[j]) { rmax[j] = acc0[j]; ridx[j] = col0; }
            if (acc1[j] > rmax[j]) { rmax[j] = acc1[j]; ridx[j] = col0 + 16; }
        }

#ifdef HAVE_TDM
        if (w == 0) __builtin_amdgcn_s_wait_tensorcnt(0);
#else
        if (mp + 1 < mPairs) {
            u32x4* s = (u32x4*)&sB[cur ^ 1][0];
#pragma unroll
            for (int i = 0; i < 8; ++i) s[t + i * THREADS] = nb[i];
        }
#endif
        __syncthreads();
        cur ^= 1;
    }

    // ---- Phase 3: reduce max/argmax across the 16 columns held per lane-half
    // (xor 1/2/4/8 never crosses the 16-lane boundary). Ties -> lowest index.
#pragma unroll
    for (int off = 8; off >= 1; off >>= 1) {
#pragma unroll
        for (int j = 0; j < 8; ++j) {
            const float ov = __shfl_xor(rmax[j], off, 32);
            const int   oi = __shfl_xor(ridx[j], off, 32);
            if (ov > rmax[j] || (ov == rmax[j] && oi < ridx[j])) {
                rmax[j] = ov; ridx[j] = oi;
            }
        }
    }
    // C/D layout: lanes 0-15 hold rows 0..7 (vgpr j), lanes 16-31 rows 8..15
    if ((lane & 15) == 0) {
        const int rbase = w * 16 + ((lane >> 4) << 3);
#pragma unroll
        for (int j = 0; j < 8; ++j) {
            const int r = rbase + j;
            const float nrm = fmaxf(sNorm[r], EPSN);
            sIdx[r] = (rmax[j] > THRESH * nrm) ? ridx[j] : -1;
        }
    }
    __syncthreads();

    // ---- Phase 4: masked gather of raw memory rows, coalesced store
    float* ob = out + (size_t)b * C_DIM * HW + n0;
    for (int e = t; e < ROWS_PER_WG * C_DIM; e += THREADS) {
        const int r   = e & (ROWS_PER_WG - 1);
        const int c   = e >> 7;
        const int idx = sIdx[r];
        ob[(size_t)c * HW + r] = (idx >= 0) ? mem[(size_t)idx * C_DIM + c] : 0.f;
    }
}

// ---------------------------------------------------------------------------
extern "C" void kernel_launch(void* const* d_in, const int* in_sizes, int n_in,
                              void* d_out, int out_size, void* d_ws, size_t ws_size,
                              hipStream_t stream) {
    const float* x   = (const float*)d_in[0];   // [32, 512, 64, 64]
    const float* mem = (const float*)d_in[1];   // [1024, 512]
    float* out       = (float*)d_out;
    _Float16* bws    = (_Float16*)d_ws;         // M*512 f16 = 1 MB scratch

    const int M      = in_sizes[1] / C_DIM;     // 1024
    const int rows   = in_sizes[0] / C_DIM;     // 131072 = B*HW
    const int mTiles = M / 16;                  // 64
    const int blocks = rows / ROWS_PER_WG;      // 1024

    hm_norm_swizzle_mem<<<M, 64, 0, stream>>>(mem, bws);
    hm_gemm_argmax_gather<<<blocks, THREADS, 0, stream>>>(x, mem, bws, out, mTiles);
}
